// DynamicGNN_2482491097616
// MI455X (gfx1250) — compile-verified, matrix operand order checked by reading, not run on hardware
//
#include <hip/hip_runtime.h>
#include <hip/hip_bf16.h>

// ---------------- problem constants ----------------
constexpr int N      = 1536;
constexpr int D      = 8;
constexpr int BSTEPS = 8;
constexpr int OBS    = 33;
constexpr int HID    = 64;
constexpr int HEADS  = 4;
constexpr int CD     = 8;       // out_dim per head
constexpr int E      = N * D;   // 12288

typedef __attribute__((ext_vector_type(16))) _Float16 v16h;
typedef __attribute__((ext_vector_type(8)))  float    v8f;
typedef __attribute__((ext_vector_type(4)))  float    v4f;   // native vector for NT stores

// ---------------- WMMA helpers (CDNA5 16x16x32 f16 -> f32) ----------------
__device__ __forceinline__ int kmap(int j, int half) {
  // ISA 7.12.2: 16-bit A/B fragment, lane holds K = 8*half + j (j<8), +16 for upper VGPRs
  return (j < 8) ? (8 * half + j) : (8 + 8 * half + j);
}

__device__ __forceinline__ v8f wmma16(v16h a, v16h b, v8f c) {
  return __builtin_amdgcn_wmma_f32_16x16x32_f16(false, a, false, b, (short)0, c,
                                                false, false);
}

// Scalar, bounds-guarded A fragment (used for unaligned strides, e.g. OBS=33)
__device__ __forceinline__ v16h frag_a_rm(const float* A, int row0, int lda,
                                          int k0, int kmaxv) {
  int lane = threadIdx.x & 31;
  int m = lane & 15, half = lane >> 4;
  const float* rp = A + (size_t)(row0 + m) * lda;
  v16h f;
#pragma unroll
  for (int j = 0; j < 16; ++j) {
    int k = k0 + kmap(j, half);
    f[j] = (k < kmaxv) ? (_Float16)rp[k] : (_Float16)0.0f;
  }
  return f;
}

// Pack two contiguous float4 pairs into a fragment (lane's K run is contiguous)
__device__ __forceinline__ v16h pack_frag(v4f x0, v4f x1, v4f x2, v4f x3) {
  v16h f;
#pragma unroll
  for (int j = 0; j < 4; ++j) {
    f[j]      = (_Float16)x0[j];
    f[4 + j]  = (_Float16)x1[j];
    f[8 + j]  = (_Float16)x2[j];
    f[12 + j] = (_Float16)x3[j];
  }
  return f;
}

// Vectorized A fragment: requires 16B-aligned rows (lda % 4 == 0, k0 % 8 == 0)
// Per ISA layout: lane reads K = k0+8*half+{0..7} and K+16 -> two b128 pairs.
__device__ __forceinline__ v16h frag_a_vec(const float* A, int row0, int lda,
                                           int k0) {
  int lane = threadIdx.x & 31;
  int m = lane & 15, half = lane >> 4;
  const float* rp = A + (size_t)(row0 + m) * lda + k0 + 8 * half;
  v4f x0 = *(const v4f*)(rp + 0);
  v4f x1 = *(const v4f*)(rp + 4);
  v4f x2 = *(const v4f*)(rp + 16);
  v4f x3 = *(const v4f*)(rp + 20);
  return pack_frag(x0, x1, x2, x3);
}

// Vectorized fragment from one 16B-aligned row pointer (edge gathers)
__device__ __forceinline__ v16h frag_row_vec(const float* rowp, int k0) {
  int half = (threadIdx.x >> 4) & 1;
  const float* rp = rowp + k0 + 8 * half;
  v4f x0 = *(const v4f*)(rp + 0);
  v4f x1 = *(const v4f*)(rp + 4);
  v4f x2 = *(const v4f*)(rp + 16);
  v4f x3 = *(const v4f*)(rp + 20);
  return pack_frag(x0, x1, x2, x3);
}

// B fragment from row-major K x Nw weight matrix W[k*ldb + n] (k-major: scalar)
__device__ __forceinline__ v16h frag_b_rm(const float* W, int col0, int ldb,
                                          int k0, int kmaxv) {
  int lane = threadIdx.x & 31;
  int n = lane & 15, half = lane >> 4;
  v16h f;
#pragma unroll
  for (int j = 0; j < 16; ++j) {
    int k = k0 + kmap(j, half);
    f[j] = (k < kmaxv) ? (_Float16)W[(size_t)k * ldb + col0 + n] : (_Float16)0.0f;
  }
  return f;
}

__device__ __forceinline__ void atomicMaxF(float* addr, float val) {
  unsigned int* ua = (unsigned int*)addr;
  unsigned int old = *ua;
  while (true) {
    float f = __uint_as_float(old);
    if (f >= val) break;
    unsigned int assumed = old;
    old = atomicCAS(ua, assumed, __float_as_uint(val));
    if (old == assumed) break;
  }
}

// ---------------- kernels ----------------

// Non-temporal wide zero fill: the dense (B,4,N,N) output is 302MB > 192MB L2,
// so stream it past the cache (th:NT) and let only the sparse scatter cache.
__global__ void zero_f4_nt(v4f* __restrict__ p, long n4) {
  long i = (long)blockIdx.x * blockDim.x + threadIdx.x;
  long stride = (long)gridDim.x * blockDim.x;
  v4f z = {0.f, 0.f, 0.f, 0.f};
  for (; i < n4; i += stride) __builtin_nontemporal_store(z, &p[i]);
}

__global__ void zero_f4(v4f* __restrict__ p, long n4) {
  long i = (long)blockIdx.x * blockDim.x + threadIdx.x;
  long stride = (long)gridDim.x * blockDim.x;
  v4f z = {0.f, 0.f, 0.f, 0.f};
  for (; i < n4; i += stride) p[i] = z;
}

// ObsEmbedding: relu(relu(H_t@We1+b)@We2+b) -> layernorm.  1 wave / 16 rows.
__global__ void obs_embed(const float* __restrict__ Ht, const float* We1,
                          const float* be1, const float* We2, const float* be2,
                          const float* lng, const float* lnb,
                          float* __restrict__ Hemb) {
  __shared__ __align__(16) float sx[16 * 32];
  __shared__ __align__(16) float sy[16 * 32];
  int row0 = blockIdx.x * 16;
  int lane = threadIdx.x & 31, n = lane & 15, half = lane >> 4;

  v16h a0 = frag_a_rm(Ht, row0, OBS, 0, OBS);
  v16h a1 = frag_a_rm(Ht, row0, OBS, 32, OBS);   // only k=32 valid
#pragma unroll
  for (int nt = 0; nt < 2; ++nt) {
    v8f c = {};
    c = wmma16(a0, frag_b_rm(We1, nt * 16, 32, 0, OBS), c);
    c = wmma16(a1, frag_b_rm(We1, nt * 16, 32, 32, OBS), c);
    float b1 = be1[nt * 16 + n];
#pragma unroll
    for (int v = 0; v < 8; ++v) {
      float t = c[v] + b1;
      sx[(v + 8 * half) * 32 + nt * 16 + n] = t > 0.f ? t : 0.f;
    }
  }
  __syncthreads();
  v16h x0 = frag_a_vec(sx, 0, 32, 0);
#pragma unroll
  for (int nt = 0; nt < 2; ++nt) {
    v8f c = {};
    c = wmma16(x0, frag_b_rm(We2, nt * 16, 32, 0, 32), c);
    float b2 = be2[nt * 16 + n];
#pragma unroll
    for (int v = 0; v < 8; ++v) {
      float t = c[v] + b2;
      sy[(v + 8 * half) * 32 + nt * 16 + n] = t > 0.f ? t : 0.f;
    }
  }
  __syncthreads();
  if (lane < 16) {
    int r = lane;
    float mu = 0.f;
    for (int j = 0; j < 32; ++j) mu += sy[r * 32 + j];
    mu *= (1.f / 32.f);
    float var = 0.f;
    for (int j = 0; j < 32; ++j) {
      float d = sy[r * 32 + j] - mu;
      var += d * d;
    }
    var *= (1.f / 32.f);
    float rs = rsqrtf(var + 1e-5f);
    for (int j = 0; j < 32; ++j)
      Hemb[(size_t)(row0 + r) * 32 + j] =
          (sy[r * 32 + j] - mu) * rs * lng[j] + lnb[j];
  }
}

// GRU step: gi = e@Wih^T, gh = h@Whh^T, gates -> h_new. 1 wave / 16 rows.
// Wih is (192,32) row-major -> Wih^T fragments have the same layout as an A
// fragment over rows=cols of Wih (contiguous K runs) -> b128 loads.
__global__ void gru_step(const float* __restrict__ et, const float* __restrict__ h,
                         float* __restrict__ hnew, const float* Wih,
                         const float* Whh, const float* bih, const float* bhh) {
  __shared__ __align__(16) float sgi[16 * 192];
  __shared__ __align__(16) float sgh[16 * 192];
  int row0 = blockIdx.x * 16;
  int lane = threadIdx.x & 31, n = lane & 15, half = lane >> 4;

  v16h ae  = frag_a_vec(et, row0, 32, 0);
  v16h ah0 = frag_a_vec(h, row0, 64, 0);
  v16h ah1 = frag_a_vec(h, row0, 64, 32);

  for (int nt = 0; nt < 12; ++nt) {
    int col0 = nt * 16;
    v8f ci = {};
    ci = wmma16(ae, frag_a_vec(Wih, col0, 32, 0), ci);
    float bi = bih[col0 + n];
#pragma unroll
    for (int v = 0; v < 8; ++v) sgi[(v + 8 * half) * 192 + col0 + n] = ci[v] + bi;
    v8f ch = {};
    ch = wmma16(ah0, frag_a_vec(Whh, col0, 64, 0), ch);
    ch = wmma16(ah1, frag_a_vec(Whh, col0, 64, 32), ch);
    float bh = bhh[col0 + n];
#pragma unroll
    for (int v = 0; v < 8; ++v) sgh[(v + 8 * half) * 192 + col0 + n] = ch[v] + bh;
  }
  __syncthreads();
  for (int idx = threadIdx.x; idx < 16 * 64; idx += 32) {
    int r = idx >> 6, c = idx & 63;
    float ir = sgi[r * 192 + c], iz = sgi[r * 192 + 64 + c],
          in = sgi[r * 192 + 128 + c];
    float hr = sgh[r * 192 + c], hz = sgh[r * 192 + 64 + c],
          hn = sgh[r * 192 + 128 + c];
    float rg = 1.f / (1.f + __expf(-(ir + hr)));
    float zg = 1.f / (1.f + __expf(-(iz + hz)));
    float ng = tanhf(in + rg * hn);
    float ho = h[(size_t)(row0 + r) * 64 + c];
    hnew[(size_t)(row0 + r) * 64 + c] = (1.f - zg) * ng + zg * ho;
  }
}

// EdgeScorer: sigmoid(relu([h[src],h[dst]]@Ws1+b)@Ws2+b). 1 wave / 16 edges.
__global__ void edge_score(const float* __restrict__ h, const int* src,
                           const int* dst, const float* Ws1, const float* bs1,
                           const float* Ws2, const float* bs2,
                           float* __restrict__ score) {
  int e0 = blockIdx.x * 16;
  int lane = threadIdx.x & 31, n = lane & 15, half = lane >> 4;
  __builtin_prefetch(Ws1, 0, 0);   // global_prefetch_b8: pull weight panel

  int e = e0 + (lane & 15);
  const float* prs = h + (size_t)src[e] * 64;
  const float* prd = h + (size_t)dst[e] * 64;
  v16h a[4];
  a[0] = frag_row_vec(prs, 0);
  a[1] = frag_row_vec(prs, 32);
  a[2] = frag_row_vec(prd, 0);
  a[3] = frag_row_vec(prd, 32);

  v8f p = {};
  for (int nt = 0; nt < 4; ++nt) {
    v8f c = {};
#pragma unroll
    for (int kc = 0; kc < 4; ++kc)
      c = wmma16(a[kc], frag_b_rm(Ws1, nt * 16, 64, kc * 32, 128), c);
    float b1 = bs1[nt * 16 + n];
    float w2 = Ws2[nt * 16 + n];
#pragma unroll
    for (int v = 0; v < 8; ++v) {
      float hv = c[v] + b1;
      hv = hv > 0.f ? hv : 0.f;
      p[v] += hv * w2;
    }
  }
  // reduce over the 16 lanes of each half (columns of hidden)
#pragma unroll
  for (int off = 1; off < 16; off <<= 1) {
#pragma unroll
    for (int v = 0; v < 8; ++v) p[v] += __shfl_xor(p[v], off, 32);
  }
  if (n == 0) {
    float b2 = bs2[0];
#pragma unroll
    for (int v = 0; v < 8; ++v) {
      float s = p[v] + b2;
      score[e0 + v + 8 * half] = 1.f / (1.f + __expf(-s));
    }
  }
}

// per-source top-k over D=8 candidates (ties -> earliest index, as lax.top_k)
__global__ void topk_keep(const float* __restrict__ score, const int* kptr,
                          int* __restrict__ keep, float* __restrict__ w) {
  int row = blockIdx.x * blockDim.x + threadIdx.x;
  if (row >= N) return;
  int kk = kptr[0];
  float s[D];
#pragma unroll
  for (int j = 0; j < D; ++j) s[j] = score[row * D + j];
  unsigned chosen = 0;
  for (int t = 0; t < kk && t < D; ++t) {
    int best = -1;
    float bv = -1e38f;
#pragma unroll
    for (int j = 0; j < D; ++j)
      if (!((chosen >> j) & 1) && s[j] > bv) { bv = s[j]; best = j; }
    if (best >= 0) chosen |= (1u << best);
  }
#pragma unroll
  for (int j = 0; j < D; ++j) {
    int kp = (chosen >> j) & 1;
    keep[row * D + j] = kp;
    w[row * D + j] = kp ? s[j] : 0.f;
  }
}

// GAT features: xh = h@Wgat (N,4,8), a_s/a_d head projections. 1 wave /16 rows.
__global__ void gat_feat(const float* __restrict__ h, const float* Wgat,
                         const float* atts, const float* attd,
                         float* __restrict__ xh, float* __restrict__ a_s,
                         float* __restrict__ a_d) {
  __shared__ __align__(16) float sx[16 * 32];
  int row0 = blockIdx.x * 16;
  int lane = threadIdx.x & 31, n = lane & 15, half = lane >> 4;
  v16h a0 = frag_a_vec(h, row0, 64, 0);
  v16h a1 = frag_a_vec(h, row0, 64, 32);
#pragma unroll
  for (int nt = 0; nt < 2; ++nt) {
    v8f c = {};
    c = wmma16(a0, frag_b_rm(Wgat, nt * 16, 32, 0, 64), c);
    c = wmma16(a1, frag_b_rm(Wgat, nt * 16, 32, 32, 64), c);
#pragma unroll
    for (int v = 0; v < 8; ++v) {
      float val = c[v];
      sx[(v + 8 * half) * 32 + nt * 16 + n] = val;
      xh[(size_t)(row0 + v + 8 * half) * 32 + nt * 16 + n] = val;
    }
  }
  __syncthreads();
  for (int idx = threadIdx.x; idx < 16 * HEADS; idx += 32) {
    int r = idx >> 2, hh = idx & 3;
    float ss = 0.f, dd = 0.f;
#pragma unroll
    for (int c2 = 0; c2 < CD; ++c2) {
      float x = sx[r * 32 + hh * CD + c2];
      ss += x * atts[hh * CD + c2];
      dd += x * attd[hh * CD + c2];
    }
    a_s[(row0 + r) * HEADS + hh] = ss;
    a_d[(row0 + r) * HEADS + hh] = dd;
  }
}

// out-slice bias init + per-node softmax state init
__global__ void step_init(float* __restrict__ out_b, const float* bgat,
                          float* m, float* ssum, float* rowsum) {
  int idx = blockIdx.x * blockDim.x + threadIdx.x;
  if (idx < N * 32) out_b[idx] = bgat[idx & 31];
  if (idx < N * HEADS) {
    m[idx] = -1e30f;
    ssum[idx] = 0.f;
    rowsum[idx] = 0.f;
  }
}

__global__ void logit_max(const int* src, const int* dst, const int* keep,
                          const float* a_s, const float* a_d,
                          float* __restrict__ logit, float* m) {
  int idx = blockIdx.x * blockDim.x + threadIdx.x;
  if (idx >= E * HEADS) return;
  int e = idx >> 2, hh = idx & 3;
  if (!keep[e]) { logit[idx] = -1e30f; return; }
  float l = a_s[src[e] * HEADS + hh] + a_d[dst[e] * HEADS + hh];
  l = l > 0.f ? l : 0.2f * l;   // leaky_relu(0.2)
  logit[idx] = l;
  atomicMaxF(&m[dst[e] * HEADS + hh], l);
}

__global__ void ex_sum(const int* dst, const int* keep, const float* logit,
                       const float* m, float* __restrict__ ex, float* ssum) {
  int idx = blockIdx.x * blockDim.x + threadIdx.x;
  if (idx >= E * HEADS) return;
  int e = idx >> 2, hh = idx & 3;
  if (!keep[e]) { ex[idx] = 0.f; return; }
  float v = __expf(logit[idx] - m[dst[e] * HEADS + hh]);
  ex[idx] = v;
  atomicAdd(&ssum[dst[e] * HEADS + hh], v);
}

__global__ void alpha_rowsum(const int* src, const int* dst, const float* ex,
                             const float* ssum, float* __restrict__ alpha,
                             float* rowsum) {
  int idx = blockIdx.x * blockDim.x + threadIdx.x;
  if (idx >= E * HEADS) return;
  int e = idx >> 2, hh = idx & 3;
  float den = ssum[dst[e] * HEADS + hh];
  den = den > 1e-16f ? den : 1e-16f;
  float a = ex[idx] / den;
  alpha[idx] = a;
  if (a != 0.f) atomicAdd(&rowsum[src[e] * HEADS + hh], a);
}

// scatter messages into out slice and normalized alpha into dense A slice
__global__ void scatter_out_attn(const int* src, const int* dst,
                                 const float* alpha, const float* rowsum,
                                 const float* xh, const float* w,
                                 float* __restrict__ out_b,
                                 float* __restrict__ attn_b) {
  int idx = blockIdx.x * blockDim.x + threadIdx.x;
  if (idx >= E * HEADS) return;
  int e = idx >> 2, hh = idx & 3;
  float a = alpha[idx];
  if (a == 0.f) return;
  int s = src[e], d = dst[e];
  float coef = a * w[e];
#pragma unroll
  for (int c2 = 0; c2 < CD; ++c2)
    atomicAdd(&out_b[d * 32 + hh * CD + c2],
              coef * xh[(size_t)s * 32 + hh * CD + c2]);
  float rs = rowsum[s * HEADS + hh];
  rs = rs > 1e-9f ? rs : 1e-9f;
  atomicAdd(&attn_b[(size_t)hh * N * N + (size_t)s * N + d], a / rs);
}

// ---------------- host driver ----------------
extern "C" void kernel_launch(void* const* d_in, const int* in_sizes, int n_in,
                              void* d_out, int out_size, void* d_ws,
                              size_t ws_size, hipStream_t stream) {
  const float* Ht  = (const float*)d_in[0];
  const int*   src = (const int*)d_in[1];
  const int*   dst = (const int*)d_in[2];
  const int*   kpt = (const int*)d_in[3];
  const float* We1 = (const float*)d_in[4];
  const float* be1 = (const float*)d_in[5];
  const float* We2 = (const float*)d_in[6];
  const float* be2 = (const float*)d_in[7];
  const float* lng = (const float*)d_in[8];
  const float* lnb = (const float*)d_in[9];
  const float* Wih = (const float*)d_in[10];
  const float* Whh = (const float*)d_in[11];
  const float* bih = (const float*)d_in[12];
  const float* bhh = (const float*)d_in[13];
  const float* Ws1 = (const float*)d_in[14];
  const float* bs1 = (const float*)d_in[15];
  const float* Ws2 = (const float*)d_in[16];
  const float* bs2 = (const float*)d_in[17];
  const float* Wgt = (const float*)d_in[18];
  const float* ats = (const float*)d_in[19];
  const float* atd = (const float*)d_in[20];
  const float* bgt = (const float*)d_in[21];

  float* out  = (float*)d_out;
  float* attn = out + (size_t)BSTEPS * N * 32;

  // workspace layout (floats)
  float* ws   = (float*)d_ws;
  float* Hemb = ws;                 ws += (size_t)BSTEPS * N * 32;
  float* h0   = ws;                 ws += (size_t)N * HID;
  float* h1   = ws;                 ws += (size_t)N * HID;
  float* scr  = ws;                 ws += E;
  float* wgt  = ws;                 ws += E;
  int*   keep = (int*)ws;           ws += E;
  float* xh   = ws;                 ws += (size_t)N * 32;
  float* a_s  = ws;                 ws += (size_t)N * HEADS;
  float* a_d  = ws;                 ws += (size_t)N * HEADS;
  float* lgt  = ws;                 ws += (size_t)E * HEADS;
  float* ex   = ws;                 ws += (size_t)E * HEADS;
  float* alp  = ws;                 ws += (size_t)E * HEADS;
  float* mmx  = ws;                 ws += (size_t)N * HEADS;
  float* ssm  = ws;                 ws += (size_t)N * HEADS;
  float* rsm  = ws;                 ws += (size_t)N * HEADS;

  // zero dense attention output (dominant HBM traffic, streamed non-temporal)
  long attn_f4 = (long)BSTEPS * HEADS * N * N / 4;
  zero_f4_nt<<<4096, 256, 0, stream>>>((v4f*)attn, attn_f4);
  zero_f4<<<96, 256, 0, stream>>>((v4f*)h0, (long)N * HID / 4);

  // ObsEmbedding over all (B,N) rows: B*N/16 = 768 wave tiles
  obs_embed<<<(BSTEPS * N) / 16, 32, 0, stream>>>(Ht, We1, be1, We2, be2, lng,
                                                  lnb, Hemb);

  float* hp = h0;
  float* hn = h1;
  for (int b = 0; b < BSTEPS; ++b) {
    float* out_b  = out + (size_t)b * N * 32;
    float* attn_b = attn + (size_t)b * HEADS * N * N;

    step_init<<<(N * 32) / 256, 256, 0, stream>>>(out_b, bgt, mmx, ssm, rsm);
    gru_step<<<N / 16, 32, 0, stream>>>(Hemb + (size_t)b * N * 32, hp, hn, Wih,
                                        Whh, bih, bhh);
    edge_score<<<E / 16, 32, 0, stream>>>(hn, src, dst, Ws1, bs1, Ws2, bs2, scr);
    topk_keep<<<N / 256, 256, 0, stream>>>(scr, kpt, keep, wgt);
    gat_feat<<<N / 16, 32, 0, stream>>>(hn, Wgt, ats, atd, xh, a_s, a_d);
    logit_max<<<(E * HEADS) / 256, 256, 0, stream>>>(src, dst, keep, a_s, a_d,
                                                     lgt, mmx);
    ex_sum<<<(E * HEADS) / 256, 256, 0, stream>>>(dst, keep, lgt, mmx, ex, ssm);
    alpha_rowsum<<<(E * HEADS) / 256, 256, 0, stream>>>(src, dst, ex, ssm, alp,
                                                        rsm);
    scatter_out_attn<<<(E * HEADS) / 256, 256, 0, stream>>>(src, dst, alp, rsm,
                                                            xh, wgt, out_b,
                                                            attn_b);
    float* t = hp; hp = hn; hn = t;
  }
}
